// PaliGemmaWithExpertModel_89326729822510
// MI455X (gfx1250) — matrix operand order repaired and checked by reading, not compiled, not run on hardware
//
#include <hip/hip_runtime.h>
#include <hip/hip_bf16.h>

// ---------------------------------------------------------------------------
// Types
// ---------------------------------------------------------------------------
typedef __bf16 bf16_t;
typedef __attribute__((ext_vector_type(16))) __bf16          v16bf;
typedef __attribute__((ext_vector_type(8)))  float           v8f;
typedef __attribute__((ext_vector_type(4)))  float           f32x4;
typedef __attribute__((ext_vector_type(8)))  unsigned short  u16x8;
typedef __attribute__((ext_vector_type(4)))  unsigned short  u16x4;

union FragBF { v16bf v; u16x8 h[2]; };

__device__ __forceinline__ unsigned short f2bf_u(float f) {
  union { float f; unsigned u; } a; a.f = f;
  unsigned r = a.u + 0x7FFFu + ((a.u >> 16) & 1u);   // round-to-nearest-even
  return (unsigned short)(r >> 16);
}
__device__ __forceinline__ bf16_t f2bf(float f) {
  unsigned short s = f2bf_u(f);
  union { unsigned short s; bf16_t b; } o; o.s = s; return o.b;
}
__device__ __forceinline__ float gelu_tanh(float x) {
  float x3 = x * x * x;
  return 0.5f * x * (1.0f + tanhf(0.7978845608028654f * (x + 0.044715f * x3)));
}

// ---------------------------------------------------------------------------
// Generic bf16 WMMA GEMM:  C[m,n] = sum_k A[m,k] * W[n,k]   (C = A * W^T)
//   A: bf16 [*, lda];  W: fp32 (converted on the fly) or bf16 [*, ldw]
//   Block tile 128x128x32; 8 wave32 waves; each wave = 2x4 grid of 16x16 WMMA.
//   Row remap (both A rows and C rows):  r -> (r/s)*St + r%s + off
//   Batched z:   ptr += (z/zDiv)*SO + (z%zDiv)*SI     (A, W, C; mask uses zo)
//   EPI: 0 = none, 1 = +resid[row,col], 2 = *scale + mask
//   OBF: store bf16 instead of fp32
//   DUAL: second weight matrix W2; out = gelu(accW) * accW2   (fused MLP)
// ---------------------------------------------------------------------------
#define BM 128
#define BN 128
#define BK 32
#define LSTR 40   // padded LDS row stride (bf16 elems) to dodge bank conflicts

template<typename WT, int EPI, bool OBF, bool DUAL>
__global__ __launch_bounds__(256) void gemm_wmma(
    const bf16_t* __restrict__ A, const WT* __restrict__ W, const WT* __restrict__ W2,
    void* __restrict__ Cv, const float* __restrict__ resid, const float* __restrict__ mask,
    int K, int lda, int ldw, int ldc,
    int sA, int StA, long offA,
    int sC, int StC, long offC,
    int zDiv, long aSO, long aSI, long wSO, long wSI, long cSO, long cSI, long mSO,
    float scale)
{
  __shared__ bf16_t lA [BM * LSTR];
  __shared__ bf16_t lB [BN * LSTR];
  __shared__ bf16_t lB2[BN * LSTR];

  const int tid = threadIdx.x;
  const int zo  = blockIdx.z / zDiv;
  const int zi  = blockIdx.z - zo * zDiv;
  const bf16_t* Ab  = A + (size_t)zo * aSO + (size_t)zi * aSI;
  const WT*     Wb  = W + (size_t)zo * wSO + (size_t)zi * wSI;
  const WT*     W2b = DUAL ? (W2 + (size_t)zo * wSO + (size_t)zi * wSI) : W;
  const int bm = blockIdx.y, bn = blockIdx.x;

  // ---- cooperative-load offsets for the A tile (128x32 bf16, 16B chunks) ----
  size_t aOff[2]; int aLds[2];
#pragma unroll
  for (int i = 0; i < 2; ++i) {
    int c = tid + i * 256;
    int row = c >> 2, col = (c & 3) * 8;
    int gr  = bm * BM + row;
    long mr = (long)(gr / sA) * StA + (gr % sA) + offA;
    aOff[i] = (size_t)mr * lda + col;
    aLds[i] = row * LSTR + col;
  }
  // ---- cooperative-load offsets for the W tile ----
  size_t wOff[4]; int wLds[4];
  if constexpr (sizeof(WT) == 4) {
#pragma unroll
    for (int i = 0; i < 4; ++i) {            // 128x32 fp32 as float4 chunks
      int c = tid + i * 256;
      int row = c >> 3, col = (c & 7) * 4;
      wOff[i] = (size_t)(bn * BN + row) * ldw + col;
      wLds[i] = row * LSTR + col;
    }
  } else {
#pragma unroll
    for (int i = 0; i < 2; ++i) {            // 128x32 bf16 as 16B chunks
      int c = tid + i * 256;
      int row = c >> 2, col = (c & 3) * 8;
      wOff[i] = (size_t)(bn * BN + row) * ldw + col;
      wLds[i] = row * LSTR + col;
    }
  }

  const int wave = tid >> 5, lane = tid & 31;
  const int half = lane >> 4, l16 = lane & 15;
  const int wM = (wave & 3) * 2;   // this wave's first 16-row tile (of 8)
  const int wN = (wave >> 2) * 4;  // this wave's first 16-col tile (of 8)

  v8f acc[2][4];
  v8f acc2[2][4];
  const v8f vz = {0.f,0.f,0.f,0.f,0.f,0.f,0.f,0.f};
#pragma unroll
  for (int mi = 0; mi < 2; ++mi)
#pragma unroll
    for (int ni = 0; ni < 4; ++ni) { acc[mi][ni] = vz; acc2[mi][ni] = vz; }

  for (int kt = 0; kt < K; kt += BK) {
    // ---- stage A tile ----
#pragma unroll
    for (int i = 0; i < 2; ++i)
      *(u16x8*)&lA[aLds[i]] = *(const u16x8*)(Ab + aOff[i] + kt);
    // ---- stage W (+W2) tile, fp32 -> bf16 converted in flight ----
    if constexpr (sizeof(WT) == 4) {
#pragma unroll
      for (int i = 0; i < 4; ++i) {
        f32x4 x = *(const f32x4*)(Wb + wOff[i] + kt);
        u16x4 p; p[0]=f2bf_u(x[0]); p[1]=f2bf_u(x[1]); p[2]=f2bf_u(x[2]); p[3]=f2bf_u(x[3]);
        *(u16x4*)&lB[wLds[i]] = p;
        if constexpr (DUAL) {
          f32x4 y = *(const f32x4*)(W2b + wOff[i] + kt);
          u16x4 q; q[0]=f2bf_u(y[0]); q[1]=f2bf_u(y[1]); q[2]=f2bf_u(y[2]); q[3]=f2bf_u(y[3]);
          *(u16x4*)&lB2[wLds[i]] = q;
        }
      }
    } else {
#pragma unroll
      for (int i = 0; i < 2; ++i)
        *(u16x8*)&lB[wLds[i]] = *(const u16x8*)(Wb + wOff[i] + kt);
    }
    __syncthreads();

    // ---- A fragments (ISA layout: lanes 0-15 K{0-7,16-23}, lanes 16-31 K{8-15,24-31}) ----
    FragBF af[2];
#pragma unroll
    for (int mi = 0; mi < 2; ++mi) {
      int r = (wM + mi) * 16 + l16;
      int base = r * LSTR + half * 8;
      af[mi].h[0] = *(const u16x8*)&lA[base];
      af[mi].h[1] = *(const u16x8*)&lA[base + 16];
    }
    // ---- B fragments (lanes 0-15 K{0-7,16-23}, lanes 16-31 K{8-15,24-31}) + WMMA ----
#pragma unroll
    for (int ni = 0; ni < 4; ++ni) {
      int r = (wN + ni) * 16 + l16;
      int base = r * LSTR + half * 8;
      FragBF bfr;
      bfr.h[0] = *(const u16x8*)&lB[base];
      bfr.h[1] = *(const u16x8*)&lB[base + 16];
#pragma unroll
      for (int mi = 0; mi < 2; ++mi)
        acc[mi][ni] = __builtin_amdgcn_wmma_f32_16x16x32_bf16(
            false, af[mi].v, false, bfr.v, (short)0, acc[mi][ni], false, false);
      if constexpr (DUAL) {
        FragBF b2;
        b2.h[0] = *(const u16x8*)&lB2[base];
        b2.h[1] = *(const u16x8*)&lB2[base + 16];
#pragma unroll
        for (int mi = 0; mi < 2; ++mi)
          acc2[mi][ni] = __builtin_amdgcn_wmma_f32_16x16x32_bf16(
              false, af[mi].v, false, b2.v, (short)0, acc2[mi][ni], false, false);
      }
    }
    __syncthreads();
  }

  // ---- epilogue: C/D layout -> lanes 0-15 M=j, lanes 16-31 M=j+8; N=l16 ----
  float*  Cf = (float*)Cv;
  bf16_t* Cb = (bf16_t*)Cv;
  size_t cbase = (size_t)zo * cSO + (size_t)zi * cSI;
#pragma unroll
  for (int mi = 0; mi < 2; ++mi) {
#pragma unroll
    for (int ni = 0; ni < 4; ++ni) {
#pragma unroll
      for (int j = 0; j < 8; ++j) {
        int m = bm * BM + (wM + mi) * 16 + half * 8 + j;
        int n = bn * BN + (wN + ni) * 16 + l16;
        float v = acc[mi][ni][j];
        if constexpr (DUAL) v = gelu_tanh(v) * acc2[mi][ni][j];
        long orow = (long)(m / sC) * StC + (m % sC) + offC;
        if constexpr (EPI == 1) v += resid[(size_t)orow * ldc + n];
        if constexpr (EPI == 2) v = v * scale + mask[(size_t)zo * mSO + (size_t)m * ldc + n];
        size_t oi = cbase + (size_t)orow * ldc + n;
        if constexpr (OBF) Cb[oi] = f2bf(v); else Cf[oi] = v;
      }
    }
  }
}

// ---------------------------------------------------------------------------
// RMSNorm (Gemma: x * rsqrt(mean(x^2)+eps) * (1+w)) -> bf16
// ---------------------------------------------------------------------------
__global__ __launch_bounds__(256) void rmsnorm_bf16_k(
    const float* __restrict__ x, const float* __restrict__ g,
    bf16_t* __restrict__ out, int W)
{
  int row = blockIdx.x;
  const float* xr = x + (size_t)row * W;
  float ss = 0.f;
  for (int c = threadIdx.x; c < W; c += 256) { float v = xr[c]; ss += v * v; }
  __shared__ float red[256];
  red[threadIdx.x] = ss; __syncthreads();
  for (int s = 128; s > 0; s >>= 1) {
    if (threadIdx.x < s) red[threadIdx.x] += red[threadIdx.x + s];
    __syncthreads();
  }
  float inv = rsqrtf(red[0] / (float)W + 1e-6f);
  bf16_t* orow = out + (size_t)row * W;
  for (int c = threadIdx.x; c < W; c += 256) orow[c] = f2bf(xr[c] * inv * (1.0f + g[c]));
}

// ---------------------------------------------------------------------------
// RoPE (theta=10000, HD=256). in/out layout [B,S,nH,256]; each thread does a
// (d, d+128) pair. Writes bf16 for the attention GEMMs.
// ---------------------------------------------------------------------------
__global__ __launch_bounds__(256) void rope_bf16_k(
    const float* __restrict__ in, const int* __restrict__ pos,
    bf16_t* __restrict__ out, int nH)
{
  long idx = (long)blockIdx.x * 256 + threadIdx.x;
  int  d  = (int)(idx & 127);
  long r  = idx >> 7;           // (b*S + t)*nH + h
  long bt = r / nH;             // b*S + t
  float p  = (float)pos[bt];
  float fr = p * __powf(10000.0f, -(float)d * (1.0f / 128.0f));
  float c = __cosf(fr), s = __sinf(fr);
  const float* base = in + r * 256;
  bf16_t* ob = out + r * 256;
  float x1 = base[d], x2 = base[d + 128];
  ob[d]       = f2bf(x1 * c - x2 * s);
  ob[d + 128] = f2bf(x2 * c + x1 * s);
}

// V [B,S,256] fp32 -> V^T [B,256,S] bf16 (so P@V uses a row-major [N,K] B operand)
__global__ __launch_bounds__(256) void vtrans_bf16_k(
    const float* __restrict__ v, bf16_t* __restrict__ vt)
{
  long idx = (long)blockIdx.x * 256 + threadIdx.x;
  int  d  = (int)(idx & 255);
  long bt = idx >> 8;
  int  t  = (int)(bt & 1023);
  int  b  = (int)(bt >> 10);
  vt[((long)b * 256 + d) * 1024 + t] = f2bf(v[idx]);
}

// Row softmax over 1024 cols (mask already folded into scores) -> bf16 probs
__global__ __launch_bounds__(256) void softmax_bf16_k(
    const float* __restrict__ sc, bf16_t* __restrict__ pr)
{
  long row = blockIdx.x;
  const float* r = sc + row * 1024;
  float vals[4]; float mx = -3.4e38f;
#pragma unroll
  for (int i = 0; i < 4; ++i) { vals[i] = r[threadIdx.x + i * 256]; mx = fmaxf(mx, vals[i]); }
  __shared__ float red[256];
  red[threadIdx.x] = mx; __syncthreads();
  for (int s = 128; s > 0; s >>= 1) {
    if (threadIdx.x < s) red[threadIdx.x] = fmaxf(red[threadIdx.x], red[threadIdx.x + s]);
    __syncthreads();
  }
  mx = red[0]; __syncthreads();
  float sum = 0.f;
#pragma unroll
  for (int i = 0; i < 4; ++i) { vals[i] = __expf(vals[i] - mx); sum += vals[i]; }
  red[threadIdx.x] = sum; __syncthreads();
  for (int s = 128; s > 0; s >>= 1) {
    if (threadIdx.x < s) red[threadIdx.x] += red[threadIdx.x + s];
    __syncthreads();
  }
  float inv = 1.0f / red[0];
  bf16_t* o = pr + row * 1024;
#pragma unroll
  for (int i = 0; i < 4; ++i) o[threadIdx.x + i * 256] = f2bf(vals[i] * inv);
}

// ---------------------------------------------------------------------------
// Host orchestration
// ---------------------------------------------------------------------------
extern "C" void kernel_launch(void* const* d_in, const int* in_sizes, int n_in,
                              void* d_out, int out_size, void* d_ws, size_t ws_size,
                              hipStream_t stream)
{
  (void)in_sizes; (void)n_in; (void)out_size; (void)ws_size;
  enum { Bb=4, Stot=1024, SP=768, SS=256, MP=3072, MS=1024,
         WP=2048, WS=1024, HH=8, HD=256, HHD=2048, MLPP=16384, MLPS=4096 };

  const float* x_pre = (const float*)d_in[0];
  const float* x_suf = (const float*)d_in[1];
  const float* amask = (const float*)d_in[2];
  const int*   pos   = (const int*)d_in[3];
  const float* wq_p  = (const float*)d_in[4];
  const float* wk_p  = (const float*)d_in[5];
  const float* wv_p  = (const float*)d_in[6];
  const float* wo_p  = (const float*)d_in[7];
  const float* ln1_p = (const float*)d_in[8];
  const float* ln2_p = (const float*)d_in[9];
  const float* wg_p  = (const float*)d_in[10];
  const float* wu_p  = (const float*)d_in[11];
  const float* wd_p  = (const float*)d_in[12];
  const float* wq_s  = (const float*)d_in[13];
  const float* wk_s  = (const float*)d_in[14];
  const float* wv_s  = (const float*)d_in[15];
  const float* wo_s  = (const float*)d_in[16];
  const float* ln1_s = (const float*)d_in[17];
  const float* ln2_s = (const float*)d_in[18];
  const float* wg_s  = (const float*)d_in[19];
  const float* wu_s  = (const float*)d_in[20];
  const float* wd_s  = (const float*)d_in[21];

  char* ws = (char*)d_ws; size_t off = 0;
  auto take = [&](size_t bytes) -> void* {
    void* p = ws + off; off = (off + bytes + 255) & ~(size_t)255; return p;
  };
  bf16_t* h_pre = (bf16_t*)take((size_t)MP * WP * 2);
  bf16_t* h_suf = (bf16_t*)take((size_t)MS * WS * 2);
  float*  qf    = (float*) take((size_t)Bb * Stot * HHD * 4);
  float*  kf    = (float*) take((size_t)Bb * Stot * HD  * 4);
  float*  vf    = (float*) take((size_t)Bb * Stot * HD  * 4);
  bf16_t* qb    = (bf16_t*)take((size_t)Bb * Stot * HHD * 2);
  bf16_t* kb    = (bf16_t*)take((size_t)Bb * Stot * HD  * 2);
  bf16_t* vt    = (bf16_t*)take((size_t)Bb * HD * Stot  * 2);
  float*  sc    = (float*) take((size_t)Bb * HH * Stot * Stot * 4);
  bf16_t* pr    = (bf16_t*)take((size_t)Bb * HH * Stot * Stot * 2);
  bf16_t* attb  = (bf16_t*)take((size_t)Bb * Stot * HHD * 2);
  float*  h1p   = (float*) take((size_t)MP * WP * 4);
  float*  h1s   = (float*) take((size_t)MS * WS * 4);
  bf16_t* h2p   = (bf16_t*)take((size_t)MP * WP * 2);
  bf16_t* h2s   = (bf16_t*)take((size_t)MS * WS * 2);
  bf16_t* hmp   = (bf16_t*)take((size_t)MP * MLPP * 2);
  bf16_t* hms   = (bf16_t*)take((size_t)MS * MLPS * 2);

  // 1) RMSNorm -> bf16 activations
  rmsnorm_bf16_k<<<MP, 256, 0, stream>>>(x_pre, ln1_p, h_pre, WP);
  rmsnorm_bf16_k<<<MS, 256, 0, stream>>>(x_suf, ln1_s, h_suf, WS);

  // 2) QKV projections into concatenated [B,S,*] layouts (row remap prefix/suffix)
  gemm_wmma<float,0,false,false><<<dim3(HHD/128, MP/128, 1), 256, 0, stream>>>(
      h_pre, wq_p, nullptr, qf, nullptr, nullptr,
      WP, WP, WP, HHD,  MP, MP, 0,  SP, Stot, 0,  1, 0,0, 0,0, 0,0, 0, 1.0f);
  gemm_wmma<float,0,false,false><<<dim3(HD/128, MP/128, 1), 256, 0, stream>>>(
      h_pre, wk_p, nullptr, kf, nullptr, nullptr,
      WP, WP, WP, HD,   MP, MP, 0,  SP, Stot, 0,  1, 0,0, 0,0, 0,0, 0, 1.0f);
  gemm_wmma<float,0,false,false><<<dim3(HD/128, MP/128, 1), 256, 0, stream>>>(
      h_pre, wv_p, nullptr, vf, nullptr, nullptr,
      WP, WP, WP, HD,   MP, MP, 0,  SP, Stot, 0,  1, 0,0, 0,0, 0,0, 0, 1.0f);
  gemm_wmma<float,0,false,false><<<dim3(HHD/128, MS/128, 1), 256, 0, stream>>>(
      h_suf, wq_s, nullptr, qf, nullptr, nullptr,
      WS, WS, WS, HHD,  MS, MS, 0,  SS, Stot, SP, 1, 0,0, 0,0, 0,0, 0, 1.0f);
  gemm_wmma<float,0,false,false><<<dim3(HD/128, MS/128, 1), 256, 0, stream>>>(
      h_suf, wk_s, nullptr, kf, nullptr, nullptr,
      WS, WS, WS, HD,   MS, MS, 0,  SS, Stot, SP, 1, 0,0, 0,0, 0,0, 0, 1.0f);
  gemm_wmma<float,0,false,false><<<dim3(HD/128, MS/128, 1), 256, 0, stream>>>(
      h_suf, wv_s, nullptr, vf, nullptr, nullptr,
      WS, WS, WS, HD,   MS, MS, 0,  SS, Stot, SP, 1, 0,0, 0,0, 0,0, 0, 1.0f);

  // 3) RoPE (q,k) and V transpose -> bf16 attention operands
  rope_bf16_k<<<(Bb * Stot * HH * 128) / 256, 256, 0, stream>>>(qf, pos, qb, HH);
  rope_bf16_k<<<(Bb * Stot * 1  * 128) / 256, 256, 0, stream>>>(kf, pos, kb, 1);
  vtrans_bf16_k<<<(Bb * Stot * HD) / 256, 256, 0, stream>>>(vf, vt);

  // 4) scores = (q @ k^T)*SCALE + mask, batched over z=(b,h), GQA via wSI=0
  gemm_wmma<bf16_t,2,false,false><<<dim3(Stot/128, Stot/128, Bb*HH), 256, 0, stream>>>(
      qb, kb, nullptr, sc, nullptr, amask,
      HD, HHD, HD, Stot,  Stot, Stot, 0,  Stot, Stot, 0,
      HH, (long)Stot*HHD, (long)HD, (long)Stot*HD, 0,
      (long)HH*Stot*Stot, (long)Stot*Stot, (long)Stot*Stot, 0.0625f);

  // 5) softmax -> bf16 probs
  softmax_bf16_k<<<Bb * HH * Stot, 256, 0, stream>>>(sc, pr);

  // 6) att = probs @ V  -> bf16 [B,S,H*HD]
  gemm_wmma<bf16_t,0,true,false><<<dim3(HD/128, Stot/128, Bb*HH), 256, 0, stream>>>(
      pr, vt, nullptr, attb, nullptr, nullptr,
      Stot, Stot, Stot, HHD,  Stot, Stot, 0,  Stot, Stot, 0,
      HH, (long)HH*Stot*Stot, (long)Stot*Stot, (long)HD*Stot, 0,
      (long)Stot*HHD, (long)HD, 0, 1.0f);

  // 7) h1 = x + att @ wo^T  (per branch, A-row remap picks branch slice)
  gemm_wmma<float,1,false,false><<<dim3(WP/128, MP/128, 1), 256, 0, stream>>>(
      attb, wo_p, nullptr, h1p, x_pre, nullptr,
      HHD, HHD, HHD, WP,  SP, Stot, 0,  MP, MP, 0,  1, 0,0, 0,0, 0,0, 0, 1.0f);
  gemm_wmma<float,1,false,false><<<dim3(WS/128, MS/128, 1), 256, 0, stream>>>(
      attb, wo_s, nullptr, h1s, x_suf, nullptr,
      HHD, HHD, HHD, WS,  SS, Stot, SP, MS, MS, 0,  1, 0,0, 0,0, 0,0, 0, 1.0f);

  // 8) second RMSNorm
  rmsnorm_bf16_k<<<MP, 256, 0, stream>>>(h1p, ln2_p, h2p, WP);
  rmsnorm_bf16_k<<<MS, 256, 0, stream>>>(h1s, ln2_s, h2s, WS);

  // 9) fused gate/up GEMM + gelu*up -> bf16 mid (DUAL mode)
  gemm_wmma<float,0,true,true><<<dim3(MLPP/128, MP/128, 1), 256, 0, stream>>>(
      h2p, wg_p, wu_p, hmp, nullptr, nullptr,
      WP, WP, WP, MLPP,  MP, MP, 0,  MP, MP, 0,  1, 0,0, 0,0, 0,0, 0, 1.0f);
  gemm_wmma<float,0,true,true><<<dim3(MLPS/128, MS/128, 1), 256, 0, stream>>>(
      h2s, wg_s, wu_s, hms, nullptr, nullptr,
      WS, WS, WS, MLPS,  MS, MS, 0,  MS, MS, 0,  1, 0,0, 0,0, 0,0, 0, 1.0f);

  // 10) down projection + residual -> fp32 outputs (concatenated in d_out)
  float* out_pre = (float*)d_out;
  float* out_suf = out_pre + (size_t)MP * WP;
  gemm_wmma<float,1,false,false><<<dim3(WP/128, MP/128, 1), 256, 0, stream>>>(
      hmp, wd_p, nullptr, out_pre, h1p, nullptr,
      MLPP, MLPP, MLPP, WP,  MP, MP, 0,  MP, MP, 0,  1, 0,0, 0,0, 0,0, 0, 1.0f);
  gemm_wmma<float,1,false,false><<<dim3(WS/128, MS/128, 1), 256, 0, stream>>>(
      hms, wd_s, nullptr, out_suf, h1s, nullptr,
      MLPS, MLPS, MLPS, WS,  MS, MS, 0,  MS, MS, 0,  1, 0,0, 0,0, 0,0, 0, 1.0f);
}
